// Seq2SeqModel_70231305224588
// MI455X (gfx1250) — matrix operand order repaired and checked by reading, not compiled, not run on hardware
//
#include <hip/hip_runtime.h>
#include <stdint.h>

typedef __bf16 bf16;
typedef __attribute__((ext_vector_type(16))) __bf16 v16bf;
typedef __attribute__((ext_vector_type(8)))  float  v8f;

#define Vv  50257
#define Ee  256
#define Hh  512
#define Bb  32
#define Ss  128
#define Tt  48
#define G3H 1536

// ---------------------------------------------------------------------------
// WMMA fragment helpers (wave32, v_wmma_f32_16x16x32_bf16)
// A 16x32 bf16: lane (m=lane&15, kh=lane>>4) holds K = kh*16 + {0..15}
// B 32x16 bf16 (fed as W rows, i.e. B^T row-major along K): lane (n=lane&15,
//   kh=lane>>4) holds K = kh*16 + {0..15}  (one contiguous 32B run)
// C/D 16x16 f32: vgpr r holds (m = r + 8*(lane>>4), n = lane&15)
// ---------------------------------------------------------------------------
__device__ __forceinline__ v16bf load_a_frag(const bf16* A, int ld, int m0, int k0) {
    int lane = threadIdx.x & 31;
    const bf16* p = A + (long)(m0 + (lane & 15)) * ld + k0 + ((lane >> 4) << 4);
    v16bf f;
#pragma unroll
    for (int j = 0; j < 16; ++j) f[j] = p[j];
    return f;
}

__device__ __forceinline__ v16bf load_b_frag(const bf16* W, int ld, int n0, int k0, int nmax) {
    int lane = threadIdx.x & 31;
    int row = n0 + (lane & 15);
    if (row >= nmax) row = nmax - 1;            // tail-tile clamp (stores are masked)
    const bf16* p = W + (long)row * ld + k0 + ((lane >> 4) << 4);
    v16bf f;
#pragma unroll
    for (int j = 0; j < 16; ++j) f[j] = p[j];
    return f;
}

__device__ __forceinline__ v8f wmma_bf16(v16bf a, v16bf b, v8f c) {
    return __builtin_amdgcn_wmma_f32_16x16x32_bf16(false, a, false, b, (short)0, c,
                                                   false, false);
}

__device__ __forceinline__ float gru_gate(float xr, float xz, float xn,
                                          float hr, float hz, float hn, float hold) {
    float rg = 1.f / (1.f + __expf(-(xr + hr)));
    float zg = 1.f / (1.f + __expf(-(xz + hz)));
    float ng = tanhf(xn + rg * hn);
    return (1.f - zg) * ng + zg * hold;
}

// ---------------------------------------------------------------------------
// Generic bf16 GEMM: C(MxN) = A(MxK) * W(NxK)^T + bias, fp32 or bf16 output.
// One wave per 16x16 tile, 8 waves (256 threads) per block. Software-pipelined
// fragment loads so WMMA overlaps the next k-step's global loads.
// ---------------------------------------------------------------------------
__global__ void __launch_bounds__(256)
gemm_bf16_kernel(const bf16* __restrict__ A, int lda,
                 const bf16* __restrict__ W, int ldw,
                 const float* __restrict__ bias,
                 float* __restrict__ Cf, bf16* __restrict__ Cb, long ldc,
                 int M, int N, int K)
{
    int wave = threadIdx.x >> 5;
    int lane = threadIdx.x & 31;
    int ntiles = (N + 15) >> 4;
    int mtiles = M >> 4;
    int tile = blockIdx.x * 8 + wave;
    if (tile >= ntiles * mtiles) return;
    int nt = tile % ntiles, mt = tile / ntiles;
    int m0 = mt << 4, n0 = nt << 4;

    v8f acc = {};
    v16bf a = load_a_frag(A, lda, m0, 0);
    v16bf b = load_b_frag(W, ldw, n0, 0, N);
    for (int k0 = 32; k0 < K; k0 += 32) {
        __builtin_prefetch(W + (long)n0 * ldw + k0 + 128, 0, 1);
        v16bf an = load_a_frag(A, lda, m0, k0);
        v16bf bn = load_b_frag(W, ldw, n0, k0, N);
        acc = wmma_bf16(a, b, acc);
        a = an; b = bn;
    }
    acc = wmma_bf16(a, b, acc);

    int n = n0 + (lane & 15);
    if (n < N) {
        float bv = bias ? bias[n] : 0.f;
#pragma unroll
        for (int r = 0; r < 8; ++r) {
            int m = m0 + r + ((lane >> 4) << 3);
            float v = acc[r] + bv;
            if (Cf) Cf[(long)m * ldc + n] = v;
            else    Cb[(long)m * ldc + n] = (bf16)v;
        }
    }
}

// ---------------------------------------------------------------------------
// M=32 GEMM (qp and the dominant vocab projection): the whole A panel
// (32 x K bf16, K<=512 -> <=32KB) is staged into LDS once per block with
// GLOBAL_LOAD_ASYNC_TO_LDS_B128 (ASYNCcnt), then all 8 waves read their A
// fragments from LDS (ds_load) while W streams from L2.
// ---------------------------------------------------------------------------
__global__ void __launch_bounds__(256)
gemm_m32_lds_kernel(const bf16* __restrict__ A,   // (32, K) row-major
                    const bf16* __restrict__ W, int ldw,
                    const float* __restrict__ bias,
                    float* __restrict__ Cf, long ldc,
                    int N, int K)
{
    __shared__ bf16 As[Bb * Hh];                      // 32KB max panel

    // Async stage A -> LDS: each thread moves 16-byte chunks.
    {
        unsigned total = (unsigned)(Bb * K * sizeof(bf16));
        for (unsigned ofs = (unsigned)threadIdx.x * 16u; ofs < total; ofs += 256u * 16u) {
            unsigned lds = (unsigned)(uintptr_t)((char*)As + ofs);
            unsigned long long g = (unsigned long long)(uintptr_t)((const char*)A + ofs);
            asm volatile("global_load_async_to_lds_b128 %0, %1, off"
                         :: "v"(lds), "v"(g) : "memory");
        }
        asm volatile("s_wait_asynccnt 0" ::: "memory");
    }
    __syncthreads();

    int wave = threadIdx.x >> 5;
    int lane = threadIdx.x & 31;
    int ntiles = (N + 15) >> 4;
    int tile = blockIdx.x * 8 + wave;
    if (tile >= ntiles * 2) return;
    int nt = tile % ntiles, mt = tile / ntiles;
    int m0 = mt << 4, n0 = nt << 4;

    v8f acc = {};
    v16bf a = load_a_frag(As, K, m0, 0);
    v16bf b = load_b_frag(W, ldw, n0, 0, N);
    for (int k0 = 32; k0 < K; k0 += 32) {
        __builtin_prefetch(W + (long)n0 * ldw + k0 + 128, 0, 1);
        v16bf an = load_a_frag(As, K, m0, k0);
        v16bf bn = load_b_frag(W, ldw, n0, k0, N);
        acc = wmma_bf16(a, b, acc);
        a = an; b = bn;
    }
    acc = wmma_bf16(a, b, acc);

    int n = n0 + (lane & 15);
    if (n < N) {
        float bv = bias ? bias[n] : 0.f;
#pragma unroll
        for (int r = 0; r < 8; ++r) {
            int m = m0 + r + ((lane >> 4) << 3);
            Cf[(long)m * ldc + n] = acc[r] + bv;
        }
    }
}

// ---------------------------------------------------------------------------
// fp32 -> bf16 (optionally column-sliced) weight conversion
// ---------------------------------------------------------------------------
__global__ void convert_slice_kernel(const float* __restrict__ in, int ldi, int coff,
                                     bf16* __restrict__ out, int ldo,
                                     int rows, int cols)
{
    long idx = (long)blockIdx.x * blockDim.x + threadIdx.x;
    if (idx >= (long)rows * cols) return;
    int c = (int)(idx % cols);
    int r = (int)(idx / cols);
    out[(long)r * ldo + c] = (bf16)in[(long)r * ldi + c + coff];
}

// X(s,b,e) = bf16(emb[src[b,s], e])
__global__ void embed_src_kernel(const int* __restrict__ src,
                                 const float* __restrict__ emb,
                                 bf16* __restrict__ X)
{
    long idx = (long)blockIdx.x * blockDim.x + threadIdx.x;
    if (idx >= (long)Ss * Bb * Ee) return;
    int e  = (int)(idx % Ee);
    int sb = (int)(idx / Ee);
    int b  = sb % Bb;
    int s  = sb / Bb;
    int tok = src[b * Ss + s];
    X[idx] = (bf16)emb[(long)tok * Ee + e];
}

// ---------------------------------------------------------------------------
// Encoder GRU scan, one persistent workgroup per direction (blockIdx.x = dir).
// h kept in LDS (fp32 + bf16 mirror). Per step: 32x1536 recurrent GEMM via
// WMMA (wave w owns columns [16w,16w+16) of each gate), fused gate math.
// ---------------------------------------------------------------------------
__global__ void __launch_bounds__(1024)
enc_scan_kernel(const bf16* __restrict__ xg_f, const bf16* __restrict__ xg_b,
                const bf16* __restrict__ Whh_f, const bf16* __restrict__ Whh_b,
                const float* __restrict__ bhh_f, const float* __restrict__ bhh_b,
                bf16* __restrict__ enc_out,   // (B,S,1024) bf16
                float* __restrict__ hT)       // (2,B,H) fp32
{
    extern __shared__ char smem[];
    float* hF = (float*)smem;                       // 32*512 f32 = 64KB
    bf16*  hB = (bf16*)(smem + Bb * Hh * 4);        // 32*512 bf16 = 32KB

    int dir = blockIdx.x;
    const bf16*  xg  = dir ? xg_b  : xg_f;
    const bf16*  Whh = dir ? Whh_b : Whh_f;
    const float* bhh = dir ? bhh_b : bhh_f;

    for (int i = threadIdx.x; i < Bb * Hh; i += 1024) { hF[i] = 0.f; hB[i] = (bf16)0.f; }
    __syncthreads();

    int lane = threadIdx.x & 31;
    int nt  = threadIdx.x >> 5;           // 0..31 -> 16-col slice of each gate
    int col = (nt << 4) + (lane & 15);
    float bR = bhh[col], bZ = bhh[512 + col], bN = bhh[1024 + col];

    for (int i = 0; i < Ss; ++i) {
        int s = dir ? (Ss - 1 - i) : i;

        v8f aR0 = {}, aZ0 = {}, aN0 = {}, aR1 = {}, aZ1 = {}, aN1 = {};
        for (int k0 = 0; k0 < Hh; k0 += 32) {
            v16bf a0 = load_a_frag(hB, Hh, 0, k0);
            v16bf a1 = load_a_frag(hB, Hh, 16, k0);
            v16bf br = load_b_frag(Whh, Hh, (nt << 4),        k0, G3H);
            v16bf bz = load_b_frag(Whh, Hh, 512 + (nt << 4),  k0, G3H);
            v16bf bn = load_b_frag(Whh, Hh, 1024 + (nt << 4), k0, G3H);
            aR0 = wmma_bf16(a0, br, aR0); aZ0 = wmma_bf16(a0, bz, aZ0); aN0 = wmma_bf16(a0, bn, aN0);
            aR1 = wmma_bf16(a1, br, aR1); aZ1 = wmma_bf16(a1, bz, aZ1); aN1 = wmma_bf16(a1, bn, aN1);
        }

        float hn0[8], hn1[8];
        const bf16* xgrow = xg + (long)s * Bb * G3H;
#pragma unroll
        for (int r = 0; r < 8; ++r) {
            int m  = r + ((lane >> 4) << 3);       // M-tile 0
            int m1 = m + 16;                       // M-tile 1
            const bf16* xm  = xgrow + (long)m  * G3H;
            const bf16* xm1 = xgrow + (long)m1 * G3H;
            hn0[r] = gru_gate((float)xm[col], (float)xm[512 + col], (float)xm[1024 + col],
                              aR0[r] + bR, aZ0[r] + bZ, aN0[r] + bN, hF[m * Hh + col]);
            hn1[r] = gru_gate((float)xm1[col], (float)xm1[512 + col], (float)xm1[1024 + col],
                              aR1[r] + bR, aZ1[r] + bZ, aN1[r] + bN, hF[m1 * Hh + col]);
        }
        __syncthreads();   // all reads of old h complete
#pragma unroll
        for (int r = 0; r < 8; ++r) {
            int m  = r + ((lane >> 4) << 3);
            int m1 = m + 16;
            hF[m * Hh + col]  = hn0[r];  hB[m * Hh + col]  = (bf16)hn0[r];
            hF[m1 * Hh + col] = hn1[r];  hB[m1 * Hh + col] = (bf16)hn1[r];
            enc_out[((long)m  * Ss + s) * 1024 + dir * 512 + col] = (bf16)hn0[r];
            enc_out[((long)m1 * Ss + s) * 1024 + dir * 512 + col] = (bf16)hn1[r];
        }
        __syncthreads();
    }

    for (int i = threadIdx.x; i < Bb * Hh; i += 1024)
        hT[(long)dir * Bb * Hh + i] = hF[i];
}

// hid[bq][c] = h_{dir}[bb][c], dir=(bq>>3)&1, bb=(bq&16)|((bq&7)<<1)
__global__ void hid_shuffle_kernel(const float* __restrict__ hT,
                                   float* __restrict__ h0f, bf16* __restrict__ h0b)
{
    int idx = blockIdx.x * blockDim.x + threadIdx.x;
    if (idx >= Bb * Hh) return;
    int bq = idx / Hh, c = idx % Hh;
    int d  = (bq >> 3) & 1;
    int bb = (bq & 0x10) | ((bq & 7) << 1);
    float v = hT[((long)d * Bb + bb) * Hh + c];
    h0f[idx] = v;
    h0b[idx] = (bf16)v;
}

// ---------------------------------------------------------------------------
// Attention + context + xdec assembly. One block per batch row, 128 threads.
// kp already contains keys_proj + attn_b (baked in as GEMM bias).
// ---------------------------------------------------------------------------
__global__ void __launch_bounds__(128)
attn_kernel(const float* __restrict__ qp,       // (B,H)
            const float* __restrict__ kp,       // (B,S,H)
            const bf16*  __restrict__ enc_out,  // (B,S,1024)
            const float* __restrict__ vW,       // (H)
            const float* __restrict__ emb,
            const int* __restrict__ src, const int* __restrict__ trg,
            int t, bf16* __restrict__ xdec)     // (B,1280)
{
    __shared__ float sc[Ss];
    __shared__ float red[128];
    int b = blockIdx.x, tid = threadIdx.x;

    {   // scores[s] = sum_h v[h] * tanh(qp[b,h] + kp[b,s,h])
        int s = tid;
        const float* q = qp + (long)b * Hh;
        const float* k = kp + ((long)b * Ss + s) * Hh;
        float acc = 0.f;
        for (int h = 0; h < Hh; ++h) acc += vW[h] * tanhf(q[h] + k[h]);
        sc[s] = acc;
    }
    __syncthreads();

    red[tid] = sc[tid]; __syncthreads();
    for (int off = 64; off; off >>= 1) {
        if (tid < off) red[tid] = fmaxf(red[tid], red[tid + off]);
        __syncthreads();
    }
    float mx = red[0]; __syncthreads();
    float ex = __expf(sc[tid] - mx);
    red[tid] = ex; __syncthreads();
    for (int off = 64; off; off >>= 1) {
        if (tid < off) red[tid] += red[tid + off];
        __syncthreads();
    }
    float inv = 1.f / red[0];
    __syncthreads();
    sc[tid] = ex * inv;
    __syncthreads();

    // ctx[k] = sum_s w[s] * enc_out[b,s,k]  ->  xdec[b, 256+k]
    for (int k = tid; k < 1024; k += 128) {
        const bf16* eo = enc_out + (long)b * Ss * 1024 + k;
        float acc = 0.f;
        for (int s = 0; s < Ss; ++s) acc += sc[s] * (float)eo[(long)s * 1024];
        xdec[(long)b * 1280 + 256 + k] = (bf16)acc;
    }

    int tok = (t == 0) ? src[b * Ss + (Ss - 1)] : trg[b * Tt + (t - 1)];
    for (int j = tid; j < Ee; j += 128)
        xdec[(long)b * 1280 + j] = (bf16)emb[(long)tok * Ee + j];
}

// ---------------------------------------------------------------------------
// Decoder GRU step: xg = xdec@Wih^T (K=1280), hg = h@Whh^T (K=512), fused
// gating. Single 1024-thread workgroup; x-gate and h-gate accumulators kept
// separate (r multiplies only hn). Double-buffered h -> no global hazards.
// ---------------------------------------------------------------------------
__global__ void __launch_bounds__(1024)
dec_gru_kernel(const bf16* __restrict__ xdec,    // (32,1280)
               const bf16* __restrict__ hcurB,   // (32,512)
               const float* __restrict__ hcurF,  // (32,512)
               const bf16* __restrict__ Wih,     // (1536,1280)
               const bf16* __restrict__ Whh,     // (1536,512)
               const float* __restrict__ bih, const float* __restrict__ bhh,
               float* __restrict__ hnxtF, bf16* __restrict__ hnxtB)
{
    int lane = threadIdx.x & 31;
    int nt  = threadIdx.x >> 5;
    int col = (nt << 4) + (lane & 15);

    v8f xR0 = {}, xZ0 = {}, xN0 = {}, xR1 = {}, xZ1 = {}, xN1 = {};
    v8f hR0 = {}, hZ0 = {}, hN0 = {}, hR1 = {}, hZ1 = {}, hN1 = {};

    for (int k0 = 0; k0 < 1280; k0 += 32) {
        v16bf a0 = load_a_frag(xdec, 1280, 0, k0);
        v16bf a1 = load_a_frag(xdec, 1280, 16, k0);
        v16bf br = load_b_frag(Wih, 1280, (nt << 4),        k0, G3H);
        v16bf bz = load_b_frag(Wih, 1280, 512 + (nt << 4),  k0, G3H);
        v16bf bn = load_b_frag(Wih, 1280, 1024 + (nt << 4), k0, G3H);
        xR0 = wmma_bf16(a0, br, xR0); xZ0 = wmma_bf16(a0, bz, xZ0); xN0 = wmma_bf16(a0, bn, xN0);
        xR1 = wmma_bf16(a1, br, xR1); xZ1 = wmma_bf16(a1, bz, xZ1); xN1 = wmma_bf16(a1, bn, xN1);
    }
    for (int k0 = 0; k0 < Hh; k0 += 32) {
        v16bf a0 = load_a_frag(hcurB, Hh, 0, k0);
        v16bf a1 = load_a_frag(hcurB, Hh, 16, k0);
        v16bf br = load_b_frag(Whh, Hh, (nt << 4),        k0, G3H);
        v16bf bz = load_b_frag(Whh, Hh, 512 + (nt << 4),  k0, G3H);
        v16bf bn = load_b_frag(Whh, Hh, 1024 + (nt << 4), k0, G3H);
        hR0 = wmma_bf16(a0, br, hR0); hZ0 = wmma_bf16(a0, bz, hZ0); hN0 = wmma_bf16(a0, bn, hN0);
        hR1 = wmma_bf16(a1, br, hR1); hZ1 = wmma_bf16(a1, bz, hZ1); hN1 = wmma_bf16(a1, bn, hN1);
    }

    float biR = bih[col], biZ = bih[512 + col], biN = bih[1024 + col];
    float bhR = bhh[col], bhZ = bhh[512 + col], bhN = bhh[1024 + col];
#pragma unroll
    for (int r = 0; r < 8; ++r) {
        int m  = r + ((lane >> 4) << 3);
        int m1 = m + 16;
        float v0 = gru_gate(xR0[r] + biR, xZ0[r] + biZ, xN0[r] + biN,
                            hR0[r] + bhR, hZ0[r] + bhZ, hN0[r] + bhN,
                            hcurF[m * Hh + col]);
        float v1 = gru_gate(xR1[r] + biR, xZ1[r] + biZ, xN1[r] + biN,
                            hR1[r] + bhR, hZ1[r] + bhZ, hN1[r] + bhN,
                            hcurF[m1 * Hh + col]);
        hnxtF[m * Hh + col]  = v0;  hnxtB[m * Hh + col]  = (bf16)v0;
        hnxtF[m1 * Hh + col] = v1;  hnxtB[m1 * Hh + col] = (bf16)v1;
    }
}

// ---------------------------------------------------------------------------
// Host launcher
// ---------------------------------------------------------------------------
extern "C" void kernel_launch(void* const* d_in, const int* in_sizes, int n_in,
                              void* d_out, int out_size, void* d_ws, size_t ws_size,
                              hipStream_t stream)
{
    (void)in_sizes; (void)n_in; (void)out_size; (void)ws_size;

    const int*   src   = (const int*)  d_in[0];
    const int*   trg   = (const int*)  d_in[1];
    const float* emb   = (const float*)d_in[2];
    const float* eWihF = (const float*)d_in[3];
    const float* eWhhF = (const float*)d_in[4];
    const float* eBihF = (const float*)d_in[5];
    const float* eBhhF = (const float*)d_in[6];
    const float* eWihB = (const float*)d_in[7];
    const float* eWhhB = (const float*)d_in[8];
    const float* eBihB = (const float*)d_in[9];
    const float* eBhhB = (const float*)d_in[10];
    const float* dWih  = (const float*)d_in[11];
    const float* dWhh  = (const float*)d_in[12];
    const float* dBih  = (const float*)d_in[13];
    const float* dBhh  = (const float*)d_in[14];
    const float* attnW = (const float*)d_in[15];
    const float* attnB = (const float*)d_in[16];
    const float* vW    = (const float*)d_in[17];
    const float* outW  = (const float*)d_in[18];
    const float* outB  = (const float*)d_in[19];
    float* preds = (float*)d_out;

    char* wsp = (char*)d_ws;
    size_t off = 0;
    auto alloc = [&](size_t bytes) -> char* {
        char* p = wsp + off;
        off += (bytes + 255) & ~(size_t)255;
        return p;
    };

    bf16*  X     = (bf16*) alloc((size_t)Ss * Bb * Ee * 2);
    bf16*  wEihF = (bf16*) alloc((size_t)G3H * Ee * 2);
    bf16*  wEihB = (bf16*) alloc((size_t)G3H * Ee * 2);
    bf16*  wEhhF = (bf16*) alloc((size_t)G3H * Hh * 2);
    bf16*  wEhhB = (bf16*) alloc((size_t)G3H * Hh * 2);
    bf16*  xgF   = (bf16*) alloc((size_t)Ss * Bb * G3H * 2);
    bf16*  xgB   = (bf16*) alloc((size_t)Ss * Bb * G3H * 2);
    bf16*  encO  = (bf16*) alloc((size_t)Bb * Ss * 1024 * 2);
    float* hT    = (float*)alloc((size_t)2 * Bb * Hh * 4);
    float* kp    = (float*)alloc((size_t)Bb * Ss * Hh * 4);
    bf16*  wDih  = (bf16*) alloc((size_t)G3H * 1280 * 2);
    bf16*  wDhh  = (bf16*) alloc((size_t)G3H * Hh * 2);
    bf16*  wQ    = (bf16*) alloc((size_t)Hh * Hh * 2);
    bf16*  wK    = (bf16*) alloc((size_t)Hh * 1024 * 2);
    bf16*  wOut  = (bf16*) alloc((size_t)Vv * Hh * 2);
    float* hF0   = (float*)alloc((size_t)Bb * Hh * 4);
    float* hF1   = (float*)alloc((size_t)Bb * Hh * 4);
    bf16*  hB0   = (bf16*) alloc((size_t)Bb * Hh * 2);
    bf16*  hB1   = (bf16*) alloc((size_t)Bb * Hh * 2);
    float* qp    = (float*)alloc((size_t)Bb * Hh * 4);
    bf16*  xdec  = (bf16*) alloc((size_t)Bb * 1280 * 2);

    float* hFbuf[2] = { hF0, hF1 };
    bf16*  hBbuf[2] = { hB0, hB1 };

    auto conv = [&](const float* in, int ldi, int coff, bf16* out, int ldo,
                    int rows, int cols) {
        long total = (long)rows * cols;
        int blocks = (int)((total + 255) / 256);
        convert_slice_kernel<<<blocks, 256, 0, stream>>>(in, ldi, coff, out, ldo, rows, cols);
    };
    auto gemm = [&](const bf16* A, int lda, const bf16* W, int ldw, const float* bias,
                    float* Cf, bf16* Cb, long ldc, int M, int N, int K) {
        int tiles = ((N + 15) / 16) * (M / 16);
        int blocks = (tiles + 7) / 8;
        gemm_bf16_kernel<<<blocks, 256, 0, stream>>>(A, lda, W, ldw, bias, Cf, Cb, ldc, M, N, K);
    };
    auto gemm_m32 = [&](const bf16* A, const bf16* W, int ldw, const float* bias,
                        float* Cf, long ldc, int N, int K) {
        int tiles = ((N + 15) / 16) * 2;
        int blocks = (tiles + 7) / 8;
        gemm_m32_lds_kernel<<<blocks, 256, 0, stream>>>(A, W, ldw, bias, Cf, ldc, N, K);
    };

    // Weight conversions to bf16
    conv(eWihF, Ee,  0,   wEihF, Ee,   G3H, Ee);
    conv(eWihB, Ee,  0,   wEihB, Ee,   G3H, Ee);
    conv(eWhhF, Hh,  0,   wEhhF, Hh,   G3H, Hh);
    conv(eWhhB, Hh,  0,   wEhhB, Hh,   G3H, Hh);
    conv(dWih,  1280, 0,  wDih,  1280, G3H, 1280);
    conv(dWhh,  Hh,  0,   wDhh,  Hh,   G3H, Hh);
    conv(attnW, G3H, 0,   wQ,    Hh,   Hh,  Hh);     // attn_W[:, :512]
    conv(attnW, G3H, Hh,  wK,    1024, Hh,  1024);   // attn_W[:, 512:]
    conv(outW,  Hh,  0,   wOut,  Hh,   Vv,  Hh);

    // Embedding gather -> X (S,B,E) bf16
    {
        long total = (long)Ss * Bb * Ee;
        embed_src_kernel<<<(int)((total + 255) / 256), 256, 0, stream>>>(src, emb, X);
    }

    // Encoder input pre-GEMMs: xg = X @ Wih^T + bih   (4096 x 1536, K=256)
    gemm(X, Ee, wEihF, Ee, eBihF, nullptr, xgF, G3H, Ss * Bb, G3H, Ee);
    gemm(X, Ee, wEihB, Ee, eBihB, nullptr, xgB, G3H, Ss * Bb, G3H, Ee);

    // Encoder recurrent scans (both directions concurrently)
    enc_scan_kernel<<<2, 1024, Bb * Hh * 4 + Bb * Hh * 2, stream>>>(
        xgF, xgB, wEhhF, wEhhB, eBhhF, eBhhB, encO, hT);

    // Initial decoder hidden
    hid_shuffle_kernel<<<(Bb * Hh + 255) / 256, 256, 0, stream>>>(hT, hF0, hB0);

    // keys_proj = enc_out @ Wk^T + attn_b   (4096 x 512, K=1024)
    gemm(encO, 1024, wK, 1024, attnB, kp, nullptr, Hh, Bb * Ss, Hh, 1024);

    // Decoder loop
    for (int t = 0; t < Tt; ++t) {
        int cur = t & 1, nxt = cur ^ 1;
        // qp = h @ Wq^T  (32 x 512, K=512), A panel staged via async->LDS
        gemm_m32(hBbuf[cur], wQ, Hh, nullptr, qp, Hh, Hh, Hh);
        // attention + context + xdec
        attn_kernel<<<Bb, 128, 0, stream>>>(qp, kp, encO, vW, emb, src, trg, t, xdec);
        // GRU step
        dec_gru_kernel<<<1, 1024, 0, stream>>>(xdec, hBbuf[cur], hFbuf[cur],
                                               wDih, wDhh, dBih, dBhh,
                                               hFbuf[nxt], hBbuf[nxt]);
        // Output projection: preds[:, t, :] = h_new @ out_W^T + out_b
        gemm_m32(hBbuf[nxt], wOut, Hh, outB,
                 preds + (size_t)t * Vv, (long)Tt * Vv, Vv, Hh);
    }
}
